// BiLSTMModule_41549513621513
// MI455X (gfx1250) — matrix operand order repaired and checked by reading, not compiled, not run on hardware
//
#include <hip/hip_runtime.h>
#include <cstdint>

typedef _Float16 h16;
typedef __attribute__((ext_vector_type(16))) _Float16 v16h;
typedef __attribute__((ext_vector_type(8)))  _Float16 v8h;
typedef __attribute__((ext_vector_type(8)))  float    v8f;

#define TT    1024
#define BB    128
#define HID   100
#define NG    400            // 4*H
#define M_TOT (TT * BB)      // 131072
#define MTILES (M_TOT / 16)  // 8192
#define NT    25             // 400/16 N-tiles
#define NKT0  4              // K-chunks layer0 input (K 100 -> 128)
#define NKT1  7              // K-chunks layer1 input (K 200 -> 224)
#define NKTH  4              // K-chunks hidden (K 100 -> 128)

// Fragment-order layouts (per 16x16x32 WMMA tile):
//  A-frag chunk [m_tile][kt][lane(32)][16]  : lane = half*16+l16, m = m_tile*16+l16,
//     elems 0..7 -> k = kt*32 + half*8 + e ; elems 8..15 -> k = kt*32 + 16 + half*8 + (e-8)
//  B-frag chunk [nt][kt][lane(32)][16]      : n' = nt*16 + l16, k order same as A-frag
//  C/D-frag chunk [m_tile][nt][lane(32)][8] : m = m_tile*16 + half*8 + r, n' = nt*16 + l16

__device__ __forceinline__ v8f vzero8f() {
  v8f z = {0.f, 0.f, 0.f, 0.f, 0.f, 0.f, 0.f, 0.f};
  return z;
}

// 16 contiguous f16 -> v16h (element order == storage order in frag layouts)
__device__ __forceinline__ v16h load_frag16(const h16* p) {
  v8h lo = *reinterpret_cast<const v8h*>(p);
  v8h hi = *reinterpret_cast<const v8h*>(p + 8);
  v16h r;
#pragma unroll
  for (int i = 0; i < 8; ++i) { r[i] = lo[i]; r[i + 8] = hi[i]; }
  return r;
}

__device__ __forceinline__ v8f wmma_f16(v16h a, v16h b, v8f c) {
  return __builtin_amdgcn_wmma_f32_16x16x32_f16(false, a, false, b, (short)0, c,
                                                false, false);
}

__device__ __forceinline__ float sigmoidf_(float x) {
  return 1.0f / (1.0f + __expf(-x));
}

// gfx1250 async global->LDS copy (16B), tracked by ASYNCcnt
__device__ __forceinline__ void async_copy16(uint32_t lds_off, const void* gptr) {
  asm volatile("global_load_async_to_lds_b128 %0, %1, off"
               :: "v"(lds_off), "v"((uint64_t)(uintptr_t)gptr)
               : "memory");
}
__device__ __forceinline__ void wait_async0() {
  asm volatile("s_wait_asynccnt 0" ::: "memory");
}

// A-frag slot for matrix element (m, k) in a layout with nkt K-chunks
__device__ __forceinline__ size_t afrag_idx(int m, int k, int nkt) {
  int kt = k >> 5, kk = k & 31;
  int half = (kk >> 3) & 1;
  int e = (kk & 7) + (kk >= 16 ? 8 : 0);
  int lane = half * 16 + (m & 15);
  return (((size_t)(m >> 4) * nkt + kt) * 32 + lane) * 16 + e;
}

// ---------------------------------------------------------------------------
// Pack x (B,100,T) f32 -> X0 in A-frag order, K padded 100->128
// ---------------------------------------------------------------------------
__global__ void k_pack_x(const float* __restrict__ x, h16* __restrict__ X0) {
  int idx = blockIdx.x * 256 + threadIdx.x;        // frag-order element index
  int e  = idx & 15;
  int ln = (idx >> 4) & 31;
  int kt = (idx >> 9) & (NKT0 - 1);
  int m_tile = idx >> 11;
  int half = ln >> 4;
  int m = m_tile * 16 + (ln & 15);
  int k = kt * 32 + (e < 8 ? half * 8 + e : 16 + half * 8 + (e - 8));
  int b = m & (BB - 1);
  int t = m >> 7;
  float v = 0.f;
  if (k < 100) v = x[((size_t)b * 100 + k) * TT + t];
  X0[idx] = (h16)v;
}

// ---------------------------------------------------------------------------
// Pack weights (dir,400,K) f32 -> B-frag order [dir][nt][kt][lane][16], f16,
// K zero-padded, gate rows permuted: n' = 4*hh + gate
// ---------------------------------------------------------------------------
__global__ void k_pack_w(const float* __restrict__ W, h16* __restrict__ Wp,
                         int K, int nkt, int total) {
  int idx = blockIdx.x * 256 + threadIdx.x;
  if (idx >= total) return;
  int e  = idx & 15;
  int ln = (idx >> 4) & 31;
  int c  = idx >> 9;
  int kt = c % nkt;
  int c2 = c / nkt;
  int nt = c2 % NT;
  int dir = c2 / NT;
  int np = nt * 16 + (ln & 15);
  int k  = kt * 32 + (e < 8 ? (ln >> 4) * 8 + e : 16 + (ln >> 4) * 8 + (e - 8));
  int n  = (np & 3) * HID + (np >> 2);             // original gate row
  float v = (k < K) ? W[((size_t)dir * NG + n) * K + k] : 0.f;
  Wp[idx] = (h16)v;
}

__global__ void k_fill0(uint32_t* __restrict__ p, int n) {
  int i = blockIdx.x * 256 + threadIdx.x;
  if (i < n) p[i] = 0u;
}

// ---------------------------------------------------------------------------
// gx (frag order) = X @ W^T + bias.  grid (MTILES/8, NT/5, 2), block 256.
// B tiles are shared by all 8 waves -> async-staged once into LDS.
// ---------------------------------------------------------------------------
__global__ __launch_bounds__(256) void k_gemm_gates(
    const h16* __restrict__ X, const h16* __restrict__ Wp,
    const float* __restrict__ bias, h16* __restrict__ gx, int nkt) {
  __shared__ h16 Bsh[5 * NKT1 * 512];              // up to 35.8 KB

  const int wave = threadIdx.x >> 5;
  const int ln   = threadIdx.x & 31;
  const int l16  = ln & 15;
  const int m_tile = blockIdx.x * 8 + wave;
  const int n0t  = blockIdx.y * 5;
  const int dir  = blockIdx.z;
  const h16* Wd  = Wp + (size_t)dir * NT * nkt * 512;
  h16* gxF       = gx + (size_t)dir * MTILES * NT * 256;

  // async-stage this block's B fragments (5 N-tiles, all K-chunks) into LDS
  {
    const char* gsrc = (const char*)(Wd + (size_t)n0t * nkt * 512);
    const uint32_t lds_base = (uint32_t)(uintptr_t)(&Bsh[0]);
    const int chunks = 5 * nkt * 64;               // 16-byte chunks
    for (int i = threadIdx.x; i < chunks; i += 256)
      async_copy16(lds_base + i * 16, gsrc + (size_t)i * 16);
  }
  wait_async0();
  __syncthreads();

  v8f acc[5];
#pragma unroll
  for (int i = 0; i < 5; ++i) acc[i] = vzero8f();

  for (int kt = 0; kt < nkt; ++kt) {
    const h16* ap = X + (((size_t)m_tile * nkt + kt) * 32 + ln) * 16;
    v16h a = load_frag16(ap);                      // coalesced 1KB/wave
#pragma unroll
    for (int nt = 0; nt < 5; ++nt) {
      const h16* bp = &Bsh[((nt * nkt + kt) * 32 + ln) * 16];
      v16h b = load_frag16(bp);                    // ds_load_b128 x2
      acc[nt] = wmma_f16(a, b, acc[nt]);
    }
  }
#pragma unroll
  for (int nt = 0; nt < 5; ++nt) {
    int ntg = n0t + nt;
    int np = ntg * 16 + l16;
    int norig = (np & 3) * HID + (np >> 2);
    float bv = bias[dir * NG + norig];
    v8h o;
#pragma unroll
    for (int r = 0; r < 8; ++r) o[r] = (h16)(acc[nt][r] + bv);
    *reinterpret_cast<v8h*>(gxF + (((size_t)m_tile * NT + ntg) * 32 + ln) * 8) = o;
  }
}

// ---------------------------------------------------------------------------
// Recurrence: one 1024-thread workgroup per direction (grid = 2).
// h kept in LDS in A-frag order; gx read as one b128/lane/job (frag order),
// next step's gx prefetched; quad-shuffle pointwise; c in LDS.
// ---------------------------------------------------------------------------
__global__ __launch_bounds__(1024) void k_lstm_rec(
    const h16* __restrict__ gx, const h16* __restrict__ whh,
    h16* __restrict__ out16, float* __restrict__ out32, int layer1) {
  __shared__ h16   hsh[8 * NKTH * 32 * 16];        // 32 KB, A-frag order
  __shared__ float cLds[BB][HID];                  // 51.2 KB

  const int tid  = threadIdx.x;
  const int wave = tid >> 5;
  const int ln   = tid & 31;
  const int half = ln >> 4;
  const int l16  = ln & 15;
  const int dir  = blockIdx.x;

  for (int i = tid; i < 8 * NKTH * 32 * 16; i += 1024) hsh[i] = (h16)0.f;
  for (int i = tid; i < BB * HID; i += 1024) (&cLds[0][0])[i] = 0.f;
  __syncthreads();

  const h16* gxF = gx + (size_t)dir * MTILES * NT * 256;
  const h16* whd = whh + (size_t)dir * NT * NKTH * 512;

  for (int s = 0; s < TT; ++s) {
    const int t = dir ? (TT - 1 - s) : s;

    v8f accJ[7];
#pragma unroll
    for (int i = 0; i < 7; ++i) {
      int j = wave + 32 * i;                       // wave-uniform
      if (j < 200) {
        int mt = j / NT, nt = j % NT;              // 8 M-tiles x 25 N-tiles
        v8f acc = vzero8f();
#pragma unroll
        for (int kt = 0; kt < NKTH; ++kt) {
          const h16* ap = &hsh[((mt * NKTH + kt) * 32 + ln) * 16];
          v16h a = load_frag16(ap);
          const h16* bp = whd + (((size_t)nt * NKTH + kt) * 32 + ln) * 16;
          v16h b = load_frag16(bp);                // L0/L2-hot, coalesced
          acc = wmma_f16(a, b, acc);
        }
        // add gx: one coalesced b128 per lane (frag order)
        v8h gv = *reinterpret_cast<const v8h*>(
            gxF + (((size_t)(t * 8 + mt) * NT + nt) * 32 + ln) * 8);
#pragma unroll
        for (int r = 0; r < 8; ++r) acc[r] += (float)gv[r];
        accJ[i] = acc;
        // prefetch next step's gx fragment for this job
        if (s + 1 < TT) {
          int t2 = dir ? (TT - 2 - s) : (s + 1);
          __builtin_prefetch(
              gxF + (((size_t)(t2 * 8 + mt) * NT + nt) * 32 + ln) * 8, 0, 1);
        }
      }
    }
    __syncthreads();  // all reads of h done before anyone rewrites it

    const int g    = ln & 3;
    const int base = ln & ~3;
#pragma unroll
    for (int i = 0; i < 7; ++i) {
      int j = wave + 32 * i;
      if (j < 200) {
        int mt = j / NT, nt = j % NT;
        int hh = nt * 4 + (l16 >> 2);              // lane-quad -> hidden unit
        v8f acc = accJ[i];
#pragma unroll
        for (int r = 0; r < 8; ++r) {
          float vi = __shfl(acc[r], base + 0, 32); // gate i
          float vf = __shfl(acc[r], base + 1, 32); // gate f
          float vg = __shfl(acc[r], base + 2, 32); // gate g
          float vo = __shfl(acc[r], base + 3, 32); // gate o
          if (g == 0) {
            int m = mt * 16 + half * 8 + r;        // batch index
            float c  = cLds[m][hh];
            float cn = sigmoidf_(vf) * c + sigmoidf_(vi) * tanhf(vg);
            float hn = sigmoidf_(vo) * tanhf(cn);
            cLds[m][hh] = cn;
            hsh[afrag_idx(m, hh, NKTH)] = (h16)hn;
            if (layer1) {
              out32[((size_t)t * BB + m) * 200 + dir * HID + hh] = hn;
            } else {
              // layer-1 input, A-frag order (nkt = 7), global m index
              out16[afrag_idx(t * BB + m, dir * HID + hh, NKT1)] = (h16)hn;
            }
          }
        }
      }
    }
    __syncthreads();  // new h visible before next step's WMMA reads
  }
}

// ---------------------------------------------------------------------------
extern "C" void kernel_launch(void* const* d_in, const int* in_sizes, int n_in,
                              void* d_out, int out_size, void* d_ws, size_t ws_size,
                              hipStream_t stream) {
  const float* x     = (const float*)d_in[0];
  const float* w_ih0 = (const float*)d_in[1];
  const float* w_hh0 = (const float*)d_in[2];
  const float* b0    = (const float*)d_in[3];
  const float* w_ih1 = (const float*)d_in[4];
  const float* w_hh1 = (const float*)d_in[5];
  const float* b1    = (const float*)d_in[6];
  float* out = (float*)d_out;

  uintptr_t p = (uintptr_t)d_ws;
  auto take = [&p](size_t bytes) -> void* {
    p = (p + 255) & ~(uintptr_t)255;
    void* r = (void*)p;
    p += bytes;
    return r;
  };
  h16* X0    = (h16*)take((size_t)MTILES * NKT0 * 512 * 2);  // 33.6 MB
  h16* X1    = (h16*)take((size_t)MTILES * NKT1 * 512 * 2);  // 58.7 MB
  h16* GX    = (h16*)take(2ull * MTILES * NT * 256 * 2);     // 209.7 MB (reused)
  h16* Wih0p = (h16*)take(2ull * NT * NKT0 * 512 * 2);
  h16* Whh0p = (h16*)take(2ull * NT * NKTH * 512 * 2);
  h16* Wih1p = (h16*)take(2ull * NT * NKT1 * 512 * 2);
  h16* Whh1p = (h16*)take(2ull * NT * NKTH * 512 * 2);

  k_pack_x<<<(MTILES * NKT0 * 512) / 256, 256, 0, stream>>>(x, X0);
  k_pack_w<<<(2 * NT * NKT0 * 512 + 255) / 256, 256, 0, stream>>>(
      w_ih0, Wih0p, 100, NKT0, 2 * NT * NKT0 * 512);
  k_pack_w<<<(2 * NT * NKTH * 512 + 255) / 256, 256, 0, stream>>>(
      w_hh0, Whh0p, 100, NKTH, 2 * NT * NKTH * 512);
  k_pack_w<<<(2 * NT * NKT1 * 512 + 255) / 256, 256, 0, stream>>>(
      w_ih1, Wih1p, 200, NKT1, 2 * NT * NKT1 * 512);
  k_pack_w<<<(2 * NT * NKTH * 512 + 255) / 256, 256, 0, stream>>>(
      w_hh1, Whh1p, 100, NKTH, 2 * NT * NKTH * 512);
  {
    int n = (int)((size_t)MTILES * NKT1 * 512 / 2);  // uint32 count
    k_fill0<<<(n + 255) / 256, 256, 0, stream>>>((uint32_t*)X1, n);
  }

  dim3 ggrid(MTILES / 8, NT / 5, 2);
  // layer 0
  k_gemm_gates<<<ggrid, 256, 0, stream>>>(X0, Wih0p, b0, GX, NKT0);
  k_lstm_rec<<<2, 1024, 0, stream>>>(GX, Whh0p, X1, nullptr, 0);
  // layer 1
  k_gemm_gates<<<ggrid, 256, 0, stream>>>(X1, Wih1p, b1, GX, NKT1);
  k_lstm_rec<<<2, 1024, 0, stream>>>(GX, Whh1p, nullptr, out, 1);

  (void)in_sizes; (void)n_in; (void)out_size; (void)ws_size;
}
